// EncoderPerceiver_25151328485618
// MI455X (gfx1250) — matrix operand order repaired and checked by reading, not compiled, not run on hardware
//
#include <hip/hip_runtime.h>
#include <math.h>

// ---------------------------------------------------------------------------
// Types for CDNA5 WMMA (gfx1250, wave32)
// ---------------------------------------------------------------------------
typedef __attribute__((ext_vector_type(16))) __bf16 v16bf;
typedef __attribute__((ext_vector_type(8)))  float  v8f;

union Frag16 { unsigned int u[8]; v16bf v; };

// ---------------------------------------------------------------------------
// Helpers
// ---------------------------------------------------------------------------
__device__ __forceinline__ unsigned short f2bf(float f) {
  union { float f; unsigned u; } v; v.f = f;
  unsigned r = v.u + 0x7FFFu + ((v.u >> 16) & 1u);   // round-to-nearest-even
  return (unsigned short)(r >> 16);
}
__device__ __forceinline__ float bf2f(unsigned short h) {
  union { unsigned u; float f; } v; v.u = ((unsigned)h) << 16;
  return v.f;
}
__device__ __forceinline__ float wred_sum(float v) {
  #pragma unroll
  for (int o = 16; o; o >>= 1) v += __shfl_xor(v, o, 32);
  return v;
}
__device__ __forceinline__ float gelu_exact(float x) {
  return 0.5f * x * (1.0f + erff(x * 0.70710678118654752f));
}

#define HW   50176
#define CHW  150528
#define DLAT 512
#define NLAT 256
#define ROWS 2048      // 8 samples * 256 latents

// ---------------------------------------------------------------------------
// K1: per-sample mean / sumsq over the 150528-long LN vector (data + posenc)
// grid (8, 16), block 256. stats[n*2] += sum, stats[n*2+1] += sumsq
// ---------------------------------------------------------------------------
__global__ void __launch_bounds__(256) k_stats(const float* __restrict__ data,
                                               float* __restrict__ stats) {
  __shared__ float rs[256], rq[256];
  const int s = blockIdx.x;
  const int beg = blockIdx.y * 9408, end = beg + 9408;   // 150528/16
  float sm = 0.f, sq = 0.f;
  for (int idx = beg + threadIdx.x; idx < end; idx += 256) {
    int c = idx / HW;
    int p = idx - c * HW;
    int h = p / 224, w = p - h * 224;
    float pe = (c == 0) ? __sinf((float)h) : (c == 1) ? __cosf((float)h)
                                                      : __sinf((float)w);
    float v = data[(size_t)s * CHW + idx] + pe;
    sm += v; sq += v * v;
  }
  rs[threadIdx.x] = sm; rq[threadIdx.x] = sq;
  __syncthreads();
  for (int o = 128; o; o >>= 1) {
    if (threadIdx.x < o) { rs[threadIdx.x] += rs[threadIdx.x + o];
                           rq[threadIdx.x] += rq[threadIdx.x + o]; }
    __syncthreads();
  }
  if (threadIdx.x == 0) {
    atomicAdd(&stats[s * 2 + 0], rs[0]);
    atomicAdd(&stats[s * 2 + 1], rq[0]);
  }
}

// ---------------------------------------------------------------------------
// K2: cross attention (flash softmax over 50176 pixels, dh=8, 1 head)
// One wave per (sample, latent). grid 256 blocks x 256 threads (8 waves).
// Writes X = latent_init + attn_out @ wo + bo   (f32, 2048x512)
// ---------------------------------------------------------------------------
__global__ void __launch_bounds__(256) k_cross(
    const float* __restrict__ data, const float* __restrict__ lat,
    const float* __restrict__ g,  const float* __restrict__ b,
    const float* __restrict__ g0, const float* __restrict__ b0,
    const float* __restrict__ wq, const float* __restrict__ bq,
    const float* __restrict__ wk, const float* __restrict__ bk,
    const float* __restrict__ wv, const float* __restrict__ bv,
    const float* __restrict__ wo, const float* __restrict__ bo,
    const float* __restrict__ stats, float* __restrict__ X) {
  const int wave = threadIdx.x >> 5, lane = threadIdx.x & 31;
  const int n = blockIdx.x >> 5;
  const int l = ((blockIdx.x & 31) << 3) + wave;

  // --- LN(latent row) and q = hn @ wq + bq (wave cooperative) ---
  float s = 0.f, ss = 0.f;
  for (int i = lane; i < DLAT; i += 32) { float x = lat[l * DLAT + i]; s += x; ss += x * x; }
  s = wred_sum(s); ss = wred_sum(ss);
  const float mean = s * (1.f / DLAT);
  const float rstd = rsqrtf(ss * (1.f / DLAT) - mean * mean + 1e-5f);
  float q[8];
  #pragma unroll
  for (int j = 0; j < 8; ++j) q[j] = 0.f;
  for (int i = lane; i < DLAT; i += 32) {
    float hn = (lat[l * DLAT + i] - mean) * rstd * g0[i] + b0[i];
    #pragma unroll
    for (int j = 0; j < 8; ++j) q[j] += hn * wq[i * 8 + j];
  }
  #pragma unroll
  for (int j = 0; j < 8; ++j) q[j] = wred_sum(q[j]) + bq[j];

  // --- data LN stats for this sample ---
  const float dm  = stats[n * 2 + 0] * (1.f / CHW);
  const float drs = rsqrtf(stats[n * 2 + 1] * (1.f / CHW) - dm * dm + 1e-5f);

  // --- preload tiny K/V projections (3x8 each) ---
  float Wk[3][8], Wv[3][8], Bk[8], Bv[8];
  #pragma unroll
  for (int c = 0; c < 3; ++c)
    #pragma unroll
    for (int j = 0; j < 8; ++j) { Wk[c][j] = wk[c * 8 + j]; Wv[c][j] = wv[c * 8 + j]; }
  #pragma unroll
  for (int j = 0; j < 8; ++j) { Bk[j] = bk[j]; Bv[j] = bv[j]; }

  // --- per-lane online softmax over pixel subset ---
  float m = -1e30f, lsum = 0.f, acc[8];
  #pragma unroll
  for (int j = 0; j < 8; ++j) acc[j] = 0.f;
  const float scale = 0.3535533905932738f;  // 1/sqrt(8)
  const float* d0 = data + (size_t)n * CHW;
  for (int p = lane; p < HW; p += 32) {
    int h = p / 224, w = p - h * 224;
    float y0 = (d0[p]            + __sinf((float)h) - dm) * drs * g[p]            + b[p];
    float y1 = (d0[HW + p]       + __cosf((float)h) - dm) * drs * g[HW + p]       + b[HW + p];
    float y2 = (d0[2 * HW + p]   + __sinf((float)w) - dm) * drs * g[2 * HW + p]   + b[2 * HW + p];
    float sdot = 0.f;
    #pragma unroll
    for (int j = 0; j < 8; ++j) {
      float kk = Bk[j] + y0 * Wk[0][j] + y1 * Wk[1][j] + y2 * Wk[2][j];
      sdot += q[j] * kk;
    }
    sdot *= scale;
    float mn = fmaxf(m, sdot);
    float c1 = __expf(m - mn), e = __expf(sdot - mn);
    lsum = lsum * c1 + e;
    #pragma unroll
    for (int j = 0; j < 8; ++j) {
      float vv = Bv[j] + y0 * Wv[0][j] + y1 * Wv[1][j] + y2 * Wv[2][j];
      acc[j] = acc[j] * c1 + e * vv;
    }
    m = mn;
  }
  // --- merge lanes (butterfly; associative online-softmax combine) ---
  #pragma unroll
  for (int o = 16; o; o >>= 1) {
    float m2 = __shfl_xor(m, o, 32), l2 = __shfl_xor(lsum, o, 32);
    float a2[8];
    #pragma unroll
    for (int j = 0; j < 8; ++j) a2[j] = __shfl_xor(acc[j], o, 32);
    float mn = fmaxf(m, m2);
    float c1 = __expf(m - mn), c2 = __expf(m2 - mn);
    lsum = lsum * c1 + l2 * c2;
    #pragma unroll
    for (int j = 0; j < 8; ++j) acc[j] = acc[j] * c1 + a2[j] * c2;
    m = mn;
  }
  const float inv = 1.f / lsum;
  float y[8];
  #pragma unroll
  for (int j = 0; j < 8; ++j) y[j] = acc[j] * inv;

  // --- output projection (8 -> 512) + residual vs latent_init ---
  for (int j = lane; j < DLAT; j += 32) {
    float o = bo[j];
    #pragma unroll
    for (int i = 0; i < 8; ++i) o += y[i] * wo[i * DLAT + j];
    X[((size_t)n * NLAT + l) * DLAT + j] = lat[l * DLAT + j] + o;
  }
}

// ---------------------------------------------------------------------------
// K3: row LayerNorm (512) -> bf16.  grid 256 x 256 (wave per row).
// ---------------------------------------------------------------------------
__global__ void __launch_bounds__(256) k_ln(const float* __restrict__ X,
                                            const float* __restrict__ g,
                                            const float* __restrict__ b,
                                            unsigned short* __restrict__ H) {
  const int wave = threadIdx.x >> 5, lane = threadIdx.x & 31;
  const int row = blockIdx.x * 8 + wave;
  const float* x = X + (size_t)row * DLAT;
  float s = 0.f, q = 0.f;
  for (int i = lane; i < DLAT; i += 32) { float v = x[i]; s += v; q += v * v; }
  s = wred_sum(s); q = wred_sum(q);
  const float m = s * (1.f / DLAT);
  const float rstd = rsqrtf(q * (1.f / DLAT) - m * m + 1e-5f);
  for (int i = lane; i < DLAT; i += 32)
    H[(size_t)row * DLAT + i] = f2bf((x[i] - m) * rstd * g[i] + b[i]);
}

// ---------------------------------------------------------------------------
// K7: WMMA bf16 GEMM, C = A(MxK,bf16) x B(KxN,f32->bf16) + bias, epilogue:
//   mode 0: OutBf = bf16(gelu(C))        (FF up-projection)
//   mode 1: Xres += C                    (FF down-projection + residual)
// Block = 256 thr (8 waves) = 4 row-waves x 2 col-waves.
// Block tile 128x64; wave tile 32x32 -> 4 independent WMMA chains per K-step
// (2 A-frags x 2 B-frags), amortizing LDS fragment traffic and dodging the
// WMMA->WMMA same-acc RAW hazard. Next B K-chunk is prefetched from HBM
// (global_prefetch_b8) to overlap the weight stream with matrix compute.
// M,N,K are multiples of 128/64/32 here -> no edge guards.
// ---------------------------------------------------------------------------
__global__ void __launch_bounds__(256) k_gemm(
    const unsigned short* __restrict__ A, int lda,
    const float* __restrict__ B, int ldb,
    const float* __restrict__ bias,
    unsigned short* __restrict__ OutBf, float* __restrict__ Xres,
    int K, int N, int mode) {
  __shared__ unsigned short As[128 * 32];   // [row][k] 128x32 bf16 (8KB)
  __shared__ unsigned short Bs[64 * 32];    // [col][k] transposed tile (4KB)

  const int tid = threadIdx.x;
  const int w = tid >> 5, lane = tid & 31;
  const int wr = w & 3, wc = w >> 2;            // 4 row-waves x 2 col-waves
  const int r = lane & 15, kh = lane >> 4;
  const int row0 = blockIdx.y * 128, col0 = blockIdx.x * 64;

  v8f acc00 = {}, acc01 = {}, acc10 = {}, acc11 = {};

  const int arow = tid >> 1;            // 0..127 : A staging row
  const int akk  = (tid & 1) << 4;      // 0,16   : 16 bf16 = 32B per thread
  const int bcol = tid >> 2;            // 0..63  : B staging col
  const int bk8  = (tid & 3) << 3;      // 0,8,16,24

  for (int k0 = 0; k0 < K; k0 += 32) {
    // prefetch next K-chunk of the weight stream into cache
    if (k0 + 32 < K)
      __builtin_prefetch(&B[(size_t)(k0 + 32 + bk8) * ldb + col0 + bcol], 0, 3);

    // stage A tile (bf16, 32B contiguous per thread)
    {
      const uint4* src = (const uint4*)&A[(size_t)(row0 + arow) * lda + k0 + akk];
      uint4* dst = (uint4*)&As[arow * 32 + akk];
      dst[0] = src[0]; dst[1] = src[1];
    }
    // stage B tile transposed + f32->bf16 convert
    #pragma unroll
    for (int j = 0; j < 8; ++j) {
      float f = B[(size_t)(k0 + bk8 + j) * ldb + col0 + bcol];
      Bs[bcol * 32 + bk8 + j] = f2bf(f);
    }
    __syncthreads();

    // A fragments: 16-bit A 16x32 layout (lanes 0-15: K lo-half; 16-31: hi)
    Frag16 a0, a1, b0, b1;
    const unsigned int* ar0 = (const unsigned int*)&As[(wr * 32 + r) * 32];
    const unsigned int* ar1 = (const unsigned int*)&As[(wr * 32 + 16 + r) * 32];
    #pragma unroll
    for (int j = 0; j < 4; ++j) {
      a0.u[j]     = ar0[kh * 4 + j];        // K = kh*8 + 2j,2j+1
      a0.u[4 + j] = ar0[8 + kh * 4 + j];    // K = 16 + kh*8 + 2j,2j+1
      a1.u[j]     = ar1[kh * 4 + j];
      a1.u[4 + j] = ar1[8 + kh * 4 + j];
    }
    // B fragments: 16-bit B 32x16 (same K order as A: kh*8+s, 16+kh*8+s)
    const unsigned int* bc0 = (const unsigned int*)&Bs[(wc * 32 + r) * 32];
    const unsigned int* bc1 = (const unsigned int*)&Bs[(wc * 32 + 16 + r) * 32];
    #pragma unroll
    for (int j = 0; j < 4; ++j) {
      b0.u[j]     = bc0[kh * 4 + j];
      b0.u[4 + j] = bc0[8 + kh * 4 + j];
      b1.u[j]     = bc1[kh * 4 + j];
      b1.u[4 + j] = bc1[8 + kh * 4 + j];
    }
    acc00 = __builtin_amdgcn_wmma_f32_16x16x32_bf16(false, a0.v, false, b0.v,
                                                    (short)0, acc00, false, false);
    acc01 = __builtin_amdgcn_wmma_f32_16x16x32_bf16(false, a0.v, false, b1.v,
                                                    (short)0, acc01, false, false);
    acc10 = __builtin_amdgcn_wmma_f32_16x16x32_bf16(false, a1.v, false, b0.v,
                                                    (short)0, acc10, false, false);
    acc11 = __builtin_amdgcn_wmma_f32_16x16x32_bf16(false, a1.v, false, b1.v,
                                                    (short)0, acc11, false, false);
    __syncthreads();
  }

  // epilogue: C/D layout -> VGPR i: lanes 0-15 row i, lanes 16-31 row i+8
  const int rhalf = (lane < 16) ? 0 : 8;
  #pragma unroll
  for (int ti = 0; ti < 2; ++ti) {
    const int rbase = row0 + wr * 32 + ti * 16 + rhalf;
    #pragma unroll
    for (int tj = 0; tj < 2; ++tj) {
      const int cc = col0 + wc * 32 + tj * 16 + r;
      const v8f& acc = (ti == 0) ? (tj == 0 ? acc00 : acc01)
                                 : (tj == 0 ? acc10 : acc11);
      const float bb = bias[cc];
      #pragma unroll
      for (int i = 0; i < 8; ++i) {
        const int rr = rbase + i;
        float v = acc[i] + bb;
        if (mode == 0) {
          OutBf[(size_t)rr * N + cc] = f2bf(gelu_exact(v));
        } else {
          Xres[(size_t)rr * N + cc] += v;
        }
      }
    }
  }
}

// ---------------------------------------------------------------------------
// K4: latent QKV projection (512 -> 48 = [q16|k16|v16]), wave per row
// ---------------------------------------------------------------------------
__global__ void __launch_bounds__(256) k_qkv(
    const unsigned short* __restrict__ H,
    const float* __restrict__ wq, const float* __restrict__ bq,
    const float* __restrict__ wk, const float* __restrict__ bk,
    const float* __restrict__ wv, const float* __restrict__ bv,
    float* __restrict__ QKV) {
  const int wave = threadIdx.x >> 5, lane = threadIdx.x & 31;
  const int row = blockIdx.x * 8 + wave;
  float acc[48];
  #pragma unroll
  for (int t = 0; t < 48; ++t) acc[t] = 0.f;
  for (int i = lane; i < DLAT; i += 32) {
    float h = bf2f(H[(size_t)row * DLAT + i]);
    #pragma unroll
    for (int j = 0; j < 16; ++j) {
      acc[j]      += h * wq[i * 16 + j];
      acc[16 + j] += h * wk[i * 16 + j];
      acc[32 + j] += h * wv[i * 16 + j];
    }
  }
  #pragma unroll
  for (int t = 0; t < 48; ++t) acc[t] = wred_sum(acc[t]);
  for (int t = lane; t < 48; t += 32) {
    float bb = (t < 16) ? bq[t] : (t < 32) ? bk[t - 16] : bv[t - 32];
    QKV[(size_t)row * 48 + t] = acc[t] + bb;
  }
}

// ---------------------------------------------------------------------------
// K5: latent self-attention, block per (sample, head). K/V in LDS.
// ---------------------------------------------------------------------------
__global__ void __launch_bounds__(256) k_attn(const float* __restrict__ QKV,
                                              float* __restrict__ Y) {
  __shared__ float ks[NLAT * 8], vs[NLAT * 8];
  const int n = blockIdx.x >> 1, h = blockIdx.x & 1;
  const int t = threadIdx.x;
  const int row = n * NLAT + t;
  #pragma unroll
  for (int j = 0; j < 8; ++j) {
    ks[t * 8 + j] = QKV[(size_t)row * 48 + 16 + h * 8 + j];
    vs[t * 8 + j] = QKV[(size_t)row * 48 + 32 + h * 8 + j];
  }
  __syncthreads();
  float q[8];
  #pragma unroll
  for (int j = 0; j < 8; ++j) q[j] = QKV[(size_t)row * 48 + h * 8 + j];
  float m = -1e30f, l = 0.f, acc[8];
  #pragma unroll
  for (int j = 0; j < 8; ++j) acc[j] = 0.f;
  const float scale = 0.3535533905932738f;
  for (int tt = 0; tt < NLAT; ++tt) {
    float s = 0.f;
    #pragma unroll
    for (int j = 0; j < 8; ++j) s += q[j] * ks[tt * 8 + j];
    s *= scale;
    float mn = fmaxf(m, s);
    float c = __expf(m - mn), e = __expf(s - mn);
    l = l * c + e;
    #pragma unroll
    for (int j = 0; j < 8; ++j) acc[j] = acc[j] * c + e * vs[tt * 8 + j];
    m = mn;
  }
  const float inv = 1.f / l;
  #pragma unroll
  for (int j = 0; j < 8; ++j) Y[(size_t)row * 16 + h * 8 + j] = acc[j] * inv;
}

// ---------------------------------------------------------------------------
// K6: attention output projection (16 -> 512) + residual. Wave per row.
// ---------------------------------------------------------------------------
__global__ void __launch_bounds__(256) k_oproj(const float* __restrict__ Y,
                                               const float* __restrict__ wo,
                                               const float* __restrict__ bo,
                                               float* __restrict__ X) {
  const int wave = threadIdx.x >> 5, lane = threadIdx.x & 31;
  const int row = blockIdx.x * 8 + wave;
  float y[16];
  #pragma unroll
  for (int i = 0; i < 16; ++i) y[i] = Y[(size_t)row * 16 + i];
  for (int j = lane; j < DLAT; j += 32) {
    float o = bo[j];
    #pragma unroll
    for (int i = 0; i < 16; ++i) o += y[i] * wo[i * DLAT + j];
    X[(size_t)row * DLAT + j] += o;
  }
}

// ---------------------------------------------------------------------------
// K8: mean over latents -> out (8 x 512)
// ---------------------------------------------------------------------------
__global__ void __launch_bounds__(256) k_mean(const float* __restrict__ X,
                                              float* __restrict__ out) {
  const int tid = blockIdx.x * 256 + threadIdx.x;    // 0..4095
  const int n = tid >> 9, j = tid & 511;
  float s = 0.f;
  for (int l = 0; l < NLAT; ++l) s += X[((size_t)(n * NLAT + l)) * DLAT + j];
  out[tid] = s * (1.f / NLAT);
}

// ---------------------------------------------------------------------------
// Host-side launch
// ---------------------------------------------------------------------------
extern "C" void kernel_launch(void* const* d_in, const int* in_sizes, int n_in,
                              void* d_out, int out_size, void* d_ws, size_t ws_size,
                              hipStream_t stream) {
  auto F = [&](int i) { return (const float*)d_in[i]; };

  // flat input order (see reference setup_inputs insertion order)
  const float* data = F(0);
  const float* lat  = F(1);
  const float* lng  = F(2);   const float* lnb  = F(3);
  const float* ln0g = F(4);   const float* ln0b = F(5);
  const float* cwq = F(6),  *cbq = F(7),  *cwk = F(8),  *cbk = F(9);
  const float* cwv = F(10), *cbv = F(11), *cwo = F(12), *cbo = F(13);
  const float* lnffg = F(14), *lnffb = F(15);
  const float* fw1 = F(16), *fb1 = F(17), *fw2 = F(18), *fb2 = F(19);

  // workspace layout
  char* w = (char*)d_ws;
  float*          stats = (float*)w;                                   // 16 f32
  float*          X     = (float*)(w + 256);                           // 2048x512 f32
  unsigned short* HBF   = (unsigned short*)(w + 256 + 4194304);        // 2048x512 bf16
  unsigned short* T     = (unsigned short*)(w + 256 + 4194304 + 2097152);        // 2048x2048 bf16
  float*          QKV   = (float*)(w + 256 + 4194304 + 2097152 + 8388608);       // 2048x48
  float*          YATT  = (float*)(w + 256 + 4194304 + 2097152 + 8388608 + 393216); // 2048x16
  (void)ws_size; (void)n_in; (void)in_sizes; (void)out_size;

  hipMemsetAsync(stats, 0, 64, stream);
  k_stats<<<dim3(8, 16), 256, 0, stream>>>(data, stats);
  k_cross<<<256, 256, 0, stream>>>(data, lat, lng, lnb, ln0g, ln0b,
                                   cwq, cbq, cwk, cbk, cwv, cbv, cwo, cbo,
                                   stats, X);

  auto ff = [&](const float* g, const float* b, const float* w1, const float* b1,
                const float* w2, const float* b2) {
    k_ln<<<256, 256, 0, stream>>>(X, g, b, HBF);
    // up: (2048x512)@(512x2048) -> gelu -> bf16 T   (block tile 128x64)
    k_gemm<<<dim3(2048 / 64, ROWS / 128), 256, 0, stream>>>(
        HBF, DLAT, w1, 2048, b1, T, nullptr, DLAT, 2048, 0);
    // down: (2048x2048)@(2048x512) + bias + residual into X
    k_gemm<<<dim3(DLAT / 64, ROWS / 128), 256, 0, stream>>>(
        T, 2048, w2, DLAT, b2, nullptr, X, 2048, DLAT, 1);
  };

  ff(lnffg, lnffb, fw1, fb1, fw2, fb2);

  for (int blk = 0; blk < 8; ++blk) {
    const int base = 20 + blk * 16;
    const float* ln1g = F(base + 0), *ln1b = F(base + 1);
    const float* awq = F(base + 2), *abq = F(base + 3);
    const float* awk = F(base + 4), *abk = F(base + 5);
    const float* awv = F(base + 6), *abv = F(base + 7);
    const float* awo = F(base + 8), *abo = F(base + 9);
    const float* ln2g = F(base + 10), *ln2b = F(base + 11);
    const float* bw1 = F(base + 12), *bb1 = F(base + 13);
    const float* bw2 = F(base + 14), *bb2 = F(base + 15);

    k_ln<<<256, 256, 0, stream>>>(X, ln1g, ln1b, HBF);
    k_qkv<<<256, 256, 0, stream>>>(HBF, awq, abq, awk, abk, awv, abv, QKV);
    k_attn<<<16, 256, 0, stream>>>(QKV, YATT);
    k_oproj<<<256, 256, 0, stream>>>(YATT, awo, abo, X);
    ff(ln2g, ln2b, bw1, bb1, bw2, bb2);
  }

  k_mean<<<16, 256, 0, stream>>>(X, (float*)d_out);
}
